// DiscreteEncoderAdvanced_45423574122534
// MI455X (gfx1250) — matrix operand order repaired and checked by reading, not compiled, not run on hardware
//
#include <hip/hip_runtime.h>
#include <stdint.h>

// ---------------- problem constants ----------------
#define N_NODES   32768
#define EMB_DIM   128
#define N_CODES   4096
#define N_GRAPHS  512
#define BN_EPS    1e-5f

// ---------------- output layout (floats) ----------------
#define OUT_CLOGIT 0
#define OUT_CGRAPH 512
#define OUT_SGRAPH 66048        // 512 + 512*128
#define OUT_LOSSC  131584
#define OUT_LOSSS  131585
#define OUT_CNODE  131586
#define OUT_SNODE  4325890      // 131586 + 32768*128

// ---------------- workspace layout (bytes) ----------------
#define WS_HC16   ((size_t)0)          // 32768*128 f16 = 8 MB
#define WS_HS16   ((size_t)8388608)
#define WS_CB16C  ((size_t)16777216)   // 4096*128 f16 = 1 MB
#define WS_CB16S  ((size_t)17825792)
#define WS_CNC    ((size_t)18874368)   // 4096 f32 code norms
#define WS_CNS    ((size_t)18890752)
#define WS_IDXC   ((size_t)18907136)   // 32768 i32
#define WS_IDXS   ((size_t)19038208)
#define WS_GSUMC  ((size_t)19169280)   // 512*128 f32
#define WS_GSUMS  ((size_t)19431424)
#define WS_GCNT   ((size_t)19693568)   // 512 f32
#define WS_LOSSC  ((size_t)19695616)
#define WS_LOSSS  ((size_t)19695620)
#define WS_ZERO_BEGIN WS_GSUMC
#define WS_ZERO_BYTES (19695624 - 19169280)

typedef __attribute__((ext_vector_type(16))) _Float16 v16h;
typedef __attribute__((ext_vector_type(8)))  _Float16 v8h;
typedef __attribute__((ext_vector_type(4)))  _Float16 v4h;
typedef __attribute__((ext_vector_type(8)))  float    v8f;
typedef __attribute__((ext_vector_type(4)))  int      v4i;

// ---------------- CDNA5 async global->LDS copy ----------------
#if __has_builtin(__builtin_amdgcn_global_load_async_to_lds_b128)
  #define ASYNC_CP_B128(gp, lp)                                              \
    __builtin_amdgcn_global_load_async_to_lds_b128(                          \
        (v4i*)(uintptr_t)(gp), (v4i*)(uintptr_t)(lp), 0, 0)
#else
  #define ASYNC_CP_B128(gp, lp)                                              \
    asm volatile("global_load_async_to_lds_b128 %0, %1, off"                 \
                 :: "v"((unsigned)(uintptr_t)(lp)), "v"((const void*)(gp))   \
                 : "memory")
#endif

#if __has_builtin(__builtin_amdgcn_s_wait_asynccnt)
  #define WAIT_ASYNC(n) __builtin_amdgcn_s_wait_asynccnt(n)
#else
  #define WAIT_ASYNC(n) asm volatile("s_wait_asynccnt %0" :: "i"(n) : "memory")
#endif

// ============================================================
// K0: h_c = nf*(0.5+0.5*s), h_s = nf*(1-0.5*s)  -> f16
// one wave per node row (32 lanes x 4 elems)
// ============================================================
__global__ __launch_bounds__(256) void vq_prep_nodes(
    const float* __restrict__ nf, const float* __restrict__ score,
    _Float16* __restrict__ hC, _Float16* __restrict__ hS) {
  const int row  = blockIdx.x * 8 + (threadIdx.x >> 5);
  const int lane = threadIdx.x & 31;
  const float s  = score[row];
  const float fc = 0.5f + 0.5f * s;       // alpha + (1-alpha)*s
  const float fs = 1.0f - 0.5f * s;       // 1 - (1-alpha)*s
  const float4 x = ((const float4*)nf)[row * 32 + lane];
  v4h c = { (_Float16)(x.x * fc), (_Float16)(x.y * fc),
            (_Float16)(x.z * fc), (_Float16)(x.w * fc) };
  v4h t = { (_Float16)(x.x * fs), (_Float16)(x.y * fs),
            (_Float16)(x.z * fs), (_Float16)(x.w * fs) };
  ((v4h*)hC)[row * 32 + lane] = c;
  ((v4h*)hS)[row * 32 + lane] = t;
}

// ============================================================
// K1: codebook f32 -> f16, plus f32 row norms. one wave per code row.
// ============================================================
__global__ __launch_bounds__(256) void vq_prep_codes(
    const float* __restrict__ cb, _Float16* __restrict__ cb16,
    float* __restrict__ cnorm) {
  const int row  = blockIdx.x * 8 + (threadIdx.x >> 5);
  const int lane = threadIdx.x & 31;
  const float4 v = ((const float4*)cb)[row * 32 + lane];
  v4h h = { (_Float16)v.x, (_Float16)v.y, (_Float16)v.z, (_Float16)v.w };
  ((v4h*)cb16)[row * 32 + lane] = h;
  float n = v.x * v.x + v.y * v.y + v.z * v.z + v.w * v.w;
  #pragma unroll
  for (int off = 16; off; off >>= 1) n += __shfl_xor(n, off, 32);
  if (lane == 0) cnorm[row] = n;
}

// ============================================================
// K2: WMMA argmin, double-buffered async LDS staging, 32 rows/wave.
// Block = 8 waves x 32-row M-tile = 256 node rows.
// Stage = 32 codebook rows (8 KB padded LDS) async-copied while the
// current stage's 16 WMMAs run.  Each B sub-tile is held in registers
// and reused by two independent accumulation chains (rows 0-15, 16-31).
// argmin key = ||c||^2 - 2*dot (row-constant ||x||^2 dropped).
// ============================================================
#define STAGE_ROWS 32
#define N_STAGES   (N_CODES / STAGE_ROWS)   // 128
#define LDS_ROW_H  136                      // 128 halves + 16B pad
#define LDS_BUF_H  (STAGE_ROWS * LDS_ROW_H) // 4352 halves / buffer

__global__ __launch_bounds__(256) void vq_argmin_wmma(
    const _Float16* __restrict__ X,      // [32768][128] f16
    const _Float16* __restrict__ CB,     // [4096][128]  f16
    const float*    __restrict__ cnorm,  // [4096]
    int*            __restrict__ idx_out) {
  __shared__ __align__(16) _Float16 smem[2 * LDS_BUF_H];

  const int tid  = threadIdx.x;
  const int lane = tid & 31;
  const int wave = tid >> 5;
  const int row0 = (blockIdx.x * 8 + wave) * 32;   // 32-row M-tile
  const int m    = lane & 15;
  const int hsel = lane >> 4;

  // ---- load A fragments for both 16-row halves of the M-tile ----
  v16h a0[4], a1[4];
  {
    const _Float16* xr0 = X + (size_t)(row0 + m) * EMB_DIM;
    const _Float16* xr1 = X + (size_t)(row0 + 16 + m) * EMB_DIM;
    #pragma unroll
    for (int c = 0; c < 4; ++c) {
      v8h lo0 = *(const v8h*)(xr0 + 32 * c + 8 * hsel);
      v8h hi0 = *(const v8h*)(xr0 + 32 * c + 16 + 8 * hsel);
      a0[c] = __builtin_shufflevector(lo0, hi0,
                                      0,1,2,3,4,5,6,7,8,9,10,11,12,13,14,15);
      v8h lo1 = *(const v8h*)(xr1 + 32 * c + 8 * hsel);
      v8h hi1 = *(const v8h*)(xr1 + 32 * c + 16 + 8 * hsel);
      a1[c] = __builtin_shufflevector(lo1, hi1,
                                      0,1,2,3,4,5,6,7,8,9,10,11,12,13,14,15);
    }
  }

  // ---- cooperative staging addresses: 256 threads x 2 B128 / stage ----
  const int ldRow = tid >> 4;            // 0..15
  const int ldSeg = (tid & 15) * 8;      // half offset within row
  const _Float16* g0 = CB + (size_t)ldRow * EMB_DIM + ldSeg;
  const _Float16* g1 = CB + (size_t)(ldRow + 16) * EMB_DIM + ldSeg;
  const int l0 = ldRow * LDS_ROW_H + ldSeg;
  const int l1 = (ldRow + 16) * LDS_ROW_H + ldSeg;

  float bd0[8], bd1[8];
  int   bn0[8], bn1[8];
  #pragma unroll
  for (int r = 0; r < 8; ++r) {
    bd0[r] = __builtin_inff(); bn0[r] = 0;
    bd1[r] = __builtin_inff(); bn1[r] = 0;
  }

  // prologue: stage 0 into buffer 0
  ASYNC_CP_B128(g0, &smem[l0]);
  ASYNC_CP_B128(g1, &smem[l1]);

  for (int t = 0; t < N_STAGES; ++t) {
    const int buf = t & 1;
    if (t + 1 < N_STAGES) {
      const size_t goff = (size_t)(t + 1) * STAGE_ROWS * EMB_DIM;
      const int boff = (buf ^ 1) * LDS_BUF_H;
      ASYNC_CP_B128(g0 + goff, &smem[boff + l0]);
      ASYNC_CP_B128(g1 + goff, &smem[boff + l1]);
      WAIT_ASYNC(2);            // stage t landed (in-order), t+1 in flight
    } else {
      WAIT_ASYNC(0);
    }
    __syncthreads();            // all waves' copies for stage t visible

    const _Float16* bbase =
        &smem[buf * LDS_BUF_H] + (size_t)m * LDS_ROW_H + 8 * hsel;
    #pragma unroll
    for (int sub = 0; sub < 2; ++sub) {
      const _Float16* bptr = bbase + sub * (16 * LDS_ROW_H);
      // hold the full B sub-tile (K=128) in registers, reuse twice
      v16h b[4];
      #pragma unroll
      for (int c = 0; c < 4; ++c) {
        v8h blo = *(const v8h*)(bptr + 32 * c);
        v8h bhi = *(const v8h*)(bptr + 32 * c + 16);
        b[c] = __builtin_shufflevector(blo, bhi,
                                       0,1,2,3,4,5,6,7,8,9,10,11,12,13,14,15);
      }
      v8f acc0 = {0.f, 0.f, 0.f, 0.f, 0.f, 0.f, 0.f, 0.f};
      v8f acc1 = {0.f, 0.f, 0.f, 0.f, 0.f, 0.f, 0.f, 0.f};
      #pragma unroll
      for (int c = 0; c < 4; ++c) {   // two independent WMMA chains
        acc0 = __builtin_amdgcn_wmma_f32_16x16x32_f16(
            false, a0[c], false, b[c], (short)0, acc0, false, false);
        acc1 = __builtin_amdgcn_wmma_f32_16x16x32_f16(
            false, a1[c], false, b[c], (short)0, acc1, false, false);
      }
      const int   n  = t * STAGE_ROWS + sub * 16 + m;
      const float cn = cnorm[n];
      #pragma unroll
      for (int r = 0; r < 8; ++r) {
        const float k0 = cn - 2.0f * acc0[r];
        if (k0 < bd0[r]) { bd0[r] = k0; bn0[r] = n; }
        const float k1 = cn - 2.0f * acc1[r];
        if (k1 < bd1[r]) { bd1[r] = k1; bn1[r] = n; }
      }
    }
    __syncthreads();            // stage-t buffer free for overwrite at t+2
  }

  // ---- reduce argmin across the 16 lanes of each half-wave ----
  #pragma unroll
  for (int off = 8; off; off >>= 1) {
    #pragma unroll
    for (int r = 0; r < 8; ++r) {
      float od; int oi;
      od = __shfl_xor(bd0[r], off, 32); oi = __shfl_xor(bn0[r], off, 32);
      if (od < bd0[r] || (od == bd0[r] && oi < bn0[r])) { bd0[r] = od; bn0[r] = oi; }
      od = __shfl_xor(bd1[r], off, 32); oi = __shfl_xor(bn1[r], off, 32);
      if (od < bd1[r] || (od == bd1[r] && oi < bn1[r])) { bd1[r] = od; bn1[r] = oi; }
    }
  }
  if ((lane & 15) == 0) {
    #pragma unroll
    for (int r = 0; r < 8; ++r) {
      idx_out[row0 + r + 8 * hsel]      = bn0[r];  // rows 0-15 of tile
      idx_out[row0 + 16 + r + 8 * hsel] = bn1[r];  // rows 16-31 of tile
    }
  }
}

// ============================================================
// K3: gather quantize (f32 codebooks), residual outputs,
// segment sums via f32 atomics, commitment-loss accumulation.
// one wave per node row.
// ============================================================
__global__ __launch_bounds__(256) void vq_finalize_nodes(
    const float* __restrict__ nf, const float* __restrict__ score,
    const int* __restrict__ batch,
    const float* __restrict__ cbC, const float* __restrict__ cbS,
    const int* __restrict__ idxC, const int* __restrict__ idxS,
    float* __restrict__ out,
    float* __restrict__ gsumC, float* __restrict__ gsumS,
    float* __restrict__ gcnt, float* __restrict__ lossC,
    float* __restrict__ lossS) {
  const int row  = blockIdx.x * 8 + (threadIdx.x >> 5);
  const int lane = threadIdx.x & 31;
  const float s  = score[row];
  const int   g  = batch[row];
  const int   ic = idxC[row];
  const int   is = idxS[row];
  const float4 x  = ((const float4*)nf)[row * 32 + lane];
  const float4 qc = ((const float4*)cbC)[ic * 32 + lane];
  const float4 qs = ((const float4*)cbS)[is * 32 + lane];
  const float fc = 0.5f + 0.5f * s, fs = 1.0f - 0.5f * s;

  float lc = 0.f, ls = 0.f;
  float cn[4], sn[4];
  const float xv[4]  = {x.x, x.y, x.z, x.w};
  const float qcv[4] = {qc.x, qc.y, qc.z, qc.w};
  const float qsv[4] = {qs.x, qs.y, qs.z, qs.w};
  #pragma unroll
  for (int k = 0; k < 4; ++k) {
    const float hc = xv[k] * fc, hs = xv[k] * fs;
    const float dc = qcv[k] - hc, ds = qsv[k] - hs;
    lc += dc * dc; ls += ds * ds;
    cn[k] = xv[k] * s + qcv[k];             // c_node_feat
    sn[k] = xv[k] * (1.0f - s) + qsv[k];    // s_node_feat
  }
  const size_t base = (size_t)row * EMB_DIM + lane * 4;
  #pragma unroll
  for (int k = 0; k < 4; ++k) {
    out[OUT_CNODE + base + k] = cn[k];
    out[OUT_SNODE + base + k] = sn[k];
    atomicAdd(&gsumC[g * EMB_DIM + lane * 4 + k], cn[k]);
    atomicAdd(&gsumS[g * EMB_DIM + lane * 4 + k], sn[k]);
  }
  #pragma unroll
  for (int off = 16; off; off >>= 1) {
    lc += __shfl_xor(lc, off, 32);
    ls += __shfl_xor(ls, off, 32);
  }
  if (lane == 0) {
    atomicAdd(lossC, lc);
    atomicAdd(lossS, ls);
    atomicAdd(&gcnt[g], 1.0f);
  }
}

// ============================================================
// K4: mean pool + classifier (Linear -> BN eval -> ReLU -> Linear).
// one block (256 thr) per graph; 256 hidden units, one per thread.
// ============================================================
__global__ __launch_bounds__(256) void vq_pool_classify(
    const float* __restrict__ gsumC, const float* __restrict__ gsumS,
    const float* __restrict__ gcnt,
    const float* __restrict__ w1, const float* __restrict__ b1,
    const float* __restrict__ gamma, const float* __restrict__ beta,
    const float* __restrict__ rm, const float* __restrict__ rv,
    const float* __restrict__ w2, const float* __restrict__ b2,
    const float* __restrict__ lossC, const float* __restrict__ lossS,
    float* __restrict__ out) {
  const int g   = blockIdx.x;
  const int tid = threadIdx.x;
  __shared__ float gf[EMB_DIM];
  __shared__ float red[256];
  const float inv = 1.0f / fmaxf(gcnt[g], 1.0f);
  if (tid < EMB_DIM) {
    const float vc = gsumC[g * EMB_DIM + tid] * inv;
    gf[tid] = vc;
    out[OUT_CGRAPH + g * EMB_DIM + tid] = vc;
    out[OUT_SGRAPH + g * EMB_DIM + tid] = gsumS[g * EMB_DIM + tid] * inv;
  }
  __syncthreads();
  float h = b1[tid];
  #pragma unroll 4
  for (int k = 0; k < EMB_DIM; ++k) h += gf[k] * w1[k * 256 + tid];
  h = (h - rm[tid]) * rsqrtf(rv[tid] + BN_EPS) * gamma[tid] + beta[tid];
  h = fmaxf(h, 0.0f);
  red[tid] = h * w2[tid];
  #pragma unroll
  for (int s = 128; s > 0; s >>= 1) {
    __syncthreads();
    if (tid < s) red[tid] += red[tid + s];
  }
  __syncthreads();
  if (tid == 0) out[OUT_CLOGIT + g] = red[0] + b2[0];
  if (g == 0 && tid == 0) {
    const float scale = 1.0f / ((float)N_NODES * (float)EMB_DIM);  // COMMIT_W=1
    out[OUT_LOSSC] = lossC[0] * scale;
    out[OUT_LOSSS] = lossS[0] * scale;
  }
}

// ============================================================
extern "C" void kernel_launch(void* const* d_in, const int* in_sizes, int n_in,
                              void* d_out, int out_size, void* d_ws, size_t ws_size,
                              hipStream_t stream) {
  (void)in_sizes; (void)n_in; (void)out_size; (void)ws_size;
  const float* node_feat = (const float*)d_in[0];
  const float* score     = (const float*)d_in[1];
  const int*   batch     = (const int*)d_in[2];
  const float* cbC       = (const float*)d_in[3];
  const float* cbS       = (const float*)d_in[4];
  const float* w1        = (const float*)d_in[5];
  const float* b1        = (const float*)d_in[6];
  const float* gamma     = (const float*)d_in[7];
  const float* beta      = (const float*)d_in[8];
  const float* rm        = (const float*)d_in[9];
  const float* rv        = (const float*)d_in[10];
  const float* w2        = (const float*)d_in[11];
  const float* b2        = (const float*)d_in[12];
  float* out = (float*)d_out;

  char* ws = (char*)d_ws;
  _Float16* hC16  = (_Float16*)(ws + WS_HC16);
  _Float16* hS16  = (_Float16*)(ws + WS_HS16);
  _Float16* cb16C = (_Float16*)(ws + WS_CB16C);
  _Float16* cb16S = (_Float16*)(ws + WS_CB16S);
  float* cnC   = (float*)(ws + WS_CNC);
  float* cnS   = (float*)(ws + WS_CNS);
  int*   idxC  = (int*)(ws + WS_IDXC);
  int*   idxS  = (int*)(ws + WS_IDXS);
  float* gsumC = (float*)(ws + WS_GSUMC);
  float* gsumS = (float*)(ws + WS_GSUMS);
  float* gcnt  = (float*)(ws + WS_GCNT);
  float* lossC = (float*)(ws + WS_LOSSC);
  float* lossS = (float*)(ws + WS_LOSSS);

  // zero accumulators (contiguous region) — capture-safe
  (void)hipMemsetAsync(ws + WS_ZERO_BEGIN, 0, WS_ZERO_BYTES, stream);

  vq_prep_nodes<<<N_NODES / 8, 256, 0, stream>>>(node_feat, score, hC16, hS16);
  vq_prep_codes<<<N_CODES / 8, 256, 0, stream>>>(cbC, cb16C, cnC);
  vq_prep_codes<<<N_CODES / 8, 256, 0, stream>>>(cbS, cb16S, cnS);

  // 8 waves/block, 32 rows/wave -> 256 rows per block
  vq_argmin_wmma<<<N_NODES / 256, 256, 0, stream>>>(hC16, cb16C, cnC, idxC);
  vq_argmin_wmma<<<N_NODES / 256, 256, 0, stream>>>(hS16, cb16S, cnS, idxS);

  vq_finalize_nodes<<<N_NODES / 8, 256, 0, stream>>>(
      node_feat, score, batch, cbC, cbS, idxC, idxS, out,
      gsumC, gsumS, gcnt, lossC, lossS);

  vq_pool_classify<<<N_GRAPHS, 256, 0, stream>>>(
      gsumC, gsumS, gcnt, w1, b1, gamma, beta, rm, rv, w2, b2,
      lossC, lossS, out);
}
